// MHAEncoder_44530220924955
// MI455X (gfx1250) — matrix operand order repaired and checked by reading, not compiled, not run on hardware
//
#include <hip/hip_runtime.h>
#include <hip/hip_bf16.h>

// MI455X / gfx1250, wave32, WMMA f32_16x16x32_f16 everywhere.
// B=2, S=2048, E=2048, H=32, HKV=8, D=64, G=4.

typedef __attribute__((ext_vector_type(16))) _Float16 v16h;
typedef __attribute__((ext_vector_type(8)))  _Float16 v8h;
typedef __attribute__((ext_vector_type(8)))  float    v8f;
typedef __attribute__((ext_vector_type(4)))  int      v4i;

#define BATCH 2
#define SLEN  2048
#define EMB   2048
#define NH    32
#define NKV   8
#define HD    64
#define GRP   4

#if __has_builtin(__builtin_amdgcn_global_load_async_to_lds_b128)
#define HAVE_ASYNC_LDS 1
#else
#define HAVE_ASYNC_LDS 0
#endif

__device__ __forceinline__ void wait_async0() {
#if __has_builtin(__builtin_amdgcn_s_wait_asynccnt)
    __builtin_amdgcn_s_wait_asynccnt(0);
#else
    asm volatile("s_wait_asynccnt 0x0" ::: "memory");
#endif
}

#if HAVE_ASYNC_LDS
__device__ __forceinline__ void async_copy_b128(const _Float16* g, _Float16* l) {
    // param0: global source (addrspace(1) v4i*), param1: LDS dest (addrspace(3) v4i*)
    __builtin_amdgcn_global_load_async_to_lds_b128(
        (__attribute__((address_space(1))) v4i*)g,
        (__attribute__((address_space(3))) v4i*)l, 0, 0);
}
#endif

// ---- A-fragment loader: 16x32 f16 tile, row-major source, lane holds row (lane&15),
// two contiguous 8-half chunks per lane per the ISA 16-bit A layout.
__device__ __forceinline__ v16h load_a_frag(const _Float16* __restrict__ row_k0, int hi) {
    v8h c0 = *(const v8h*)(row_k0 + hi * 8);
    v8h c1 = *(const v8h*)(row_k0 + 16 + hi * 8);
    v16h a;
#pragma unroll
    for (int i = 0; i < 8; ++i) { a[i] = c0[i]; a[i + 8] = c1[i]; }
    return a;
}

// =====================================================================
// Conversion / layout kernels
// =====================================================================
__global__ __launch_bounds__(256) void f32_to_f16_kernel(const float* __restrict__ in,
                                                         _Float16* __restrict__ out, int n) {
    int i = blockIdx.x * 256 + threadIdx.x;
    if (i < n) out[i] = (_Float16)in[i];
}

// W: (Nrows x Kcols) f32 -> out: (Kcols x Nrows) f16   (B-matrix layout, K-major)
__global__ __launch_bounds__(256) void transpose_f16_kernel(const float* __restrict__ W,
                                                            _Float16* __restrict__ out,
                                                            int Nrows, int Kcols) {
    int i = blockIdx.x * 256 + threadIdx.x;
    if (i >= Nrows * Kcols) return;
    int k = i / Nrows;
    int n = i - k * Nrows;
    out[i] = (_Float16)W[(size_t)n * Kcols + k];
}

// q f32 (B,S,H,64) -> qh f16 (B,H,S,64), RoPE applied, softmax scale 1/8 folded in.
__global__ __launch_bounds__(256) void rope_q_kernel(const float* __restrict__ q,
                                                     const float* __restrict__ cosT,
                                                     const float* __restrict__ sinT,
                                                     _Float16* __restrict__ qh) {
    int i = blockIdx.x * 256 + threadIdx.x;   // B*S*H*32 = 4,194,304
    int d = i & 31;
    int h = (i >> 5) & 31;
    int s = (i >> 10) & 2047;
    int b = i >> 21;
    const float* qp = q + ((size_t)(b * SLEN + s)) * EMB + h * HD;
    float v0 = qp[d], v1 = qp[d + 32];
    float c0 = cosT[s * HD + d],      s0 = sinT[s * HD + d];
    float c1 = cosT[s * HD + d + 32], s1 = sinT[s * HD + d + 32];
    float o0 = (v0 * c0 - v1 * s0) * 0.125f;   // 1/sqrt(64)
    float o1 = (v1 * c1 + v0 * s1) * 0.125f;
    _Float16* op = qh + (((size_t)(b * NH + h)) * SLEN + s) * HD;
    op[d] = (_Float16)o0;
    op[d + 32] = (_Float16)o1;
}

// k f32 (B,S,HKV,64) -> kT f16 (B,HKV,64,S), RoPE applied (transposed for B-fragments).
__global__ __launch_bounds__(256) void rope_k_kernel(const float* __restrict__ k,
                                                     const float* __restrict__ cosT,
                                                     const float* __restrict__ sinT,
                                                     _Float16* __restrict__ kT) {
    int i = blockIdx.x * 256 + threadIdx.x;   // B*S*HKV*32 = 1,048,576
    int d = i & 31;
    int h = (i >> 5) & 7;
    int s = (i >> 8) & 2047;
    int b = i >> 19;
    const float* kp = k + ((size_t)(b * SLEN + s)) * (NKV * HD) + h * HD;
    float v0 = kp[d], v1 = kp[d + 32];
    float c0 = cosT[s * HD + d],      s0 = sinT[s * HD + d];
    float c1 = cosT[s * HD + d + 32], s1 = sinT[s * HD + d + 32];
    float o0 = v0 * c0 - v1 * s0;
    float o1 = v1 * c1 + v0 * s1;
    _Float16* op = kT + ((size_t)(b * NKV + h)) * HD * SLEN;
    op[(size_t)d * SLEN + s]        = (_Float16)o0;
    op[(size_t)(d + 32) * SLEN + s] = (_Float16)o1;
}

// v f32 (B,S,HKV,64) -> vh f16 (B,HKV,S,64)
__global__ __launch_bounds__(256) void conv_v_kernel(const float* __restrict__ v,
                                                     _Float16* __restrict__ vh) {
    int i = blockIdx.x * 256 + threadIdx.x;   // B*S*HKV*64 = 2,097,152
    int d = i & 63;
    int h = (i >> 6) & 7;
    int s = (i >> 9) & 2047;
    int b = i >> 20;
    vh[(((size_t)(b * NKV + h)) * SLEN + s) * HD + d] =
        (_Float16)v[((size_t)(b * SLEN + s)) * (NKV * HD) + h * HD + d];
}

// =====================================================================
// WMMA GEMM: C(f32, MxN) = A(f16, MxK, row-major) * B(f16, KxN, row-major)
// Block = 256 threads = 8 waves. Block tile 256(M) x 64(N); wave tile 32x64.
// B tile (32x64) double-buffered in LDS via async global->LDS copies,
// shared by all 8 waves. 8 WMMAs per 32-deep k-step per wave.
// =====================================================================
__global__ __launch_bounds__(256) void gemm_wmma_kernel(const _Float16* __restrict__ A,
                                                        const _Float16* __restrict__ B,
                                                        float* __restrict__ C,
                                                        int M, int N, int K) {
    __shared__ _Float16 lds_b[2][32 * 64];
    const int tid  = threadIdx.x;
    const int wave = tid >> 5;
    const int lane = tid & 31;
    const int hi   = lane >> 4;
    const int lm   = lane & 15;
    const int n0 = blockIdx.x * 64;
    const int m0 = blockIdx.y * 256 + wave * 32;
    const size_t row0 = (size_t)(m0 + lm) * K;        // A frag 0 rows (m0..m0+15)
    const size_t row1 = (size_t)(m0 + 16 + lm) * K;   // A frag 1 rows (m0+16..m0+31)

    const int brow = tid >> 3;        // 0..31
    const int bcol = (tid & 7) * 8;   // 0..56
    const _Float16* bsrc = &B[(size_t)brow * N + n0 + bcol];  // + k0*N per step
    _Float16* lslot0 = &lds_b[0][brow * 64 + bcol];
    _Float16* lslot1 = &lds_b[1][brow * 64 + bcol];

    v8f acc[8];
#pragma unroll
    for (int c = 0; c < 8; ++c)
#pragma unroll
        for (int r = 0; r < 8; ++r) acc[c][r] = 0.0f;

#if HAVE_ASYNC_LDS
    // prologue: fill buffer 0
    async_copy_b128(bsrc, lslot0);
    wait_async0();
    __syncthreads();
    int cur = 0;
    for (int k0 = 0; k0 < K; k0 += 32) {
        // kick off next tile into the other buffer (overlaps with WMMAs below)
        if (k0 + 32 < K)
            async_copy_b128(bsrc + (size_t)(k0 + 32) * N, cur ? lslot0 : lslot1);

        v16h a0 = load_a_frag(&A[row0 + k0], hi);
        v16h a1 = load_a_frag(&A[row1 + k0], hi);
        const _Float16* lb = &lds_b[cur][lm];
#pragma unroll
        for (int c = 0; c < 4; ++c) {
            v16h b;
#pragma unroll
            for (int i = 0; i < 8; ++i) { b[i] = lb[(hi * 8 + i) * 64 + c * 16]; b[i + 8] = lb[(16 + hi * 8 + i) * 64 + c * 16]; }
            acc[c]     = __builtin_amdgcn_wmma_f32_16x16x32_f16(
                false, a0, false, b, (short)0, acc[c], false, false);
            acc[4 + c] = __builtin_amdgcn_wmma_f32_16x16x32_f16(
                false, a1, false, b, (short)0, acc[4 + c], false, false);
        }
        if (k0 + 32 < K) wait_async0();  // next buffer's LDS writes landed
        __syncthreads();                 // all waves done reading cur; next buffer visible
        cur ^= 1;
    }
#else
    for (int k0 = 0; k0 < K; k0 += 32) {
        *(v8h*)lslot0 = *(const v8h*)(bsrc + (size_t)k0 * N);
        __syncthreads();
        v16h a0 = load_a_frag(&A[row0 + k0], hi);
        v16h a1 = load_a_frag(&A[row1 + k0], hi);
        const _Float16* lb = &lds_b[0][lm];
#pragma unroll
        for (int c = 0; c < 4; ++c) {
            v16h b;
#pragma unroll
            for (int i = 0; i < 8; ++i) { b[i] = lb[(hi * 8 + i) * 64 + c * 16]; b[i + 8] = lb[(16 + hi * 8 + i) * 64 + c * 16]; }
            acc[c]     = __builtin_amdgcn_wmma_f32_16x16x32_f16(
                false, a0, false, b, (short)0, acc[c], false, false);
            acc[4 + c] = __builtin_amdgcn_wmma_f32_16x16x32_f16(
                false, a1, false, b, (short)0, acc[4 + c], false, false);
        }
        __syncthreads();
    }
#endif

    const int ccol = n0 + lm;
#pragma unroll
    for (int c = 0; c < 4; ++c) {
#pragma unroll
        for (int r = 0; r < 8; ++r) {
            C[(size_t)(m0 + hi * 8 + r) * N + ccol + c * 16]      = acc[c][r];
            C[(size_t)(m0 + 16 + hi * 8 + r) * N + ccol + c * 16] = acc[4 + c][r];
        }
    }
}

// =====================================================================
// Flash attention (non-causal, GQA). One wave per 16-query tile.
// qh (B,H,S,64), kT (B,HKV,64,S), vh (B,HKV,S,64) -> atth f16 (B,S,E)
// =====================================================================
__global__ __launch_bounds__(256) void attn_wmma_kernel(const _Float16* __restrict__ qh,
                                                        const _Float16* __restrict__ kT,
                                                        const _Float16* __restrict__ vh,
                                                        _Float16* __restrict__ atth) {
    __shared__ _Float16 lds_p[8 * 512];   // per-wave 16x32 P tile
    const int tid  = threadIdx.x;
    const int wave = tid >> 5;
    const int lane = tid & 31;
    const int hi   = lane >> 4;
    const int lm   = lane & 15;

    const int wid = blockIdx.x * 8 + wave;    // 8192 waves total
    const int qt = wid & 127;                 // S/16
    const int g  = (wid >> 7) & 3;            // G
    const int h  = (wid >> 9) & 7;            // HKV
    const int b  = wid >> 12;                 // B
    const int hq = h * GRP + g;
    const int s0 = qt * 16;

    // Q fragments (16x64 = two 16x32 A-frags), scale already folded in.
    const _Float16* qrow = qh + (((size_t)(b * NH + hq)) * SLEN + s0 + lm) * HD;
    v16h aq0 = load_a_frag(qrow, hi);
    v16h aq1 = load_a_frag(qrow + 32, hi);

    const _Float16* kbase = kT + ((size_t)(b * NKV + h)) * HD * SLEN;   // (64, S)
    const _Float16* vbase = vh + ((size_t)(b * NKV + h)) * SLEN * HD;   // (S, 64)
    _Float16* lds_w = &lds_p[wave * 512];

    float mi[8], li[8];
    v8f o[4];
#pragma unroll
    for (int r = 0; r < 8; ++r) { mi[r] = -1e30f; li[r] = 0.0f; }
#pragma unroll
    for (int c = 0; c < 4; ++c)
#pragma unroll
        for (int r = 0; r < 8; ++r) o[c][r] = 0.0f;

    for (int t0 = 0; t0 < SLEN; t0 += 32) {
        // ---- scores: 16x32 = Q(16x64) . K^T(64x32), 4 WMMAs
        v8f sc0, sc1;
#pragma unroll
        for (int r = 0; r < 8; ++r) { sc0[r] = 0.0f; sc1[r] = 0.0f; }
        {
            v16h bk00, bk01, bk10, bk11;
#pragma unroll
            for (int i = 0; i < 8; ++i) {
                int ka = hi * 8 + i, kb = 16 + hi * 8 + i;
                bk00[i] = kbase[(size_t)ka * SLEN + t0 + lm];        bk00[i + 8] = kbase[(size_t)kb * SLEN + t0 + lm];
                bk01[i] = kbase[(size_t)ka * SLEN + t0 + 16 + lm];   bk01[i + 8] = kbase[(size_t)kb * SLEN + t0 + 16 + lm];
                bk10[i] = kbase[(size_t)(32 + ka) * SLEN + t0 + lm]; bk10[i + 8] = kbase[(size_t)(32 + kb) * SLEN + t0 + lm];
                bk11[i] = kbase[(size_t)(32 + ka) * SLEN + t0 + 16 + lm]; bk11[i + 8] = kbase[(size_t)(32 + kb) * SLEN + t0 + 16 + lm];
            }
            sc0 = __builtin_amdgcn_wmma_f32_16x16x32_f16(false, aq0, false, bk00, (short)0, sc0, false, false);
            sc0 = __builtin_amdgcn_wmma_f32_16x16x32_f16(false, aq1, false, bk10, (short)0, sc0, false, false);
            sc1 = __builtin_amdgcn_wmma_f32_16x16x32_f16(false, aq0, false, bk01, (short)0, sc1, false, false);
            sc1 = __builtin_amdgcn_wmma_f32_16x16x32_f16(false, aq1, false, bk11, (short)0, sc1, false, false);
        }

        // ---- online softmax (row reductions across 16-lane halves)
#pragma unroll
        for (int r = 0; r < 8; ++r) {
            float vmax = fmaxf(sc0[r], sc1[r]);
            vmax = fmaxf(vmax, __shfl_xor(vmax, 8, 32));
            vmax = fmaxf(vmax, __shfl_xor(vmax, 4, 32));
            vmax = fmaxf(vmax, __shfl_xor(vmax, 2, 32));
            vmax = fmaxf(vmax, __shfl_xor(vmax, 1, 32));
            float mnew  = fmaxf(mi[r], vmax);
            float alpha = __expf(mi[r] - mnew);
            float p0 = __expf(sc0[r] - mnew);
            float p1 = __expf(sc1[r] - mnew);
            float rs = p0 + p1;
            rs += __shfl_xor(rs, 8, 32);
            rs += __shfl_xor(rs, 4, 32);
            rs += __shfl_xor(rs, 2, 32);
            rs += __shfl_xor(rs, 1, 32);
            li[r] = li[r] * alpha + rs;
            mi[r] = mnew;
#pragma unroll
            for (int c = 0; c < 4; ++c) o[c][r] *= alpha;
            // C-layout -> LDS (row, key) for A-layout reload
            int row = r + hi * 8;
            lds_w[row * 32 + lm]      = (_Float16)p0;
            lds_w[row * 32 + 16 + lm] = (_Float16)p1;
        }

        // ---- P (16x32, A-layout from LDS) . V (32x64), 4 WMMAs
        v16h ap = load_a_frag(&lds_w[lm * 32], hi);
        const _Float16* vt = vbase + (size_t)t0 * HD;
#pragma unroll
        for (int c = 0; c < 4; ++c) {
            v16h bv;
#pragma unroll
            for (int i = 0; i < 8; ++i) { bv[i] = vt[(hi * 8 + i) * HD + c * 16 + lm]; bv[i + 8] = vt[(16 + hi * 8 + i) * HD + c * 16 + lm]; }
            o[c] = __builtin_amdgcn_wmma_f32_16x16x32_f16(
                false, ap, false, bv, (short)0, o[c], false, false);
        }
    }

    // ---- normalize + write f16 output at (b, s, hq*64 + d)
#pragma unroll
    for (int r = 0; r < 8; ++r) {
        float inv = 1.0f / li[r];
        int row = s0 + r + hi * 8;
#pragma unroll
        for (int c = 0; c < 4; ++c)
            atth[((size_t)(b * SLEN) + row) * EMB + hq * HD + c * 16 + lm] =
                (_Float16)(o[c][r] * inv);
    }
}

// =====================================================================
// Host-side launch
// =====================================================================
extern "C" void kernel_launch(void* const* d_in, const int* in_sizes, int n_in,
                              void* d_out, int out_size, void* d_ws, size_t ws_size,
                              hipStream_t stream) {
    (void)in_sizes; (void)n_in; (void)out_size; (void)ws_size;
    const float* x    = (const float*)d_in[0];
    const float* cosT = (const float*)d_in[1];
    const float* sinT = (const float*)d_in[2];
    const float* Wq   = (const float*)d_in[3];
    const float* Wk   = (const float*)d_in[4];
    const float* Wv   = (const float*)d_in[5];
    const float* Wo   = (const float*)d_in[6];
    float* out = (float*)d_out;

    const size_t MS = (size_t)BATCH * SLEN;          // 4096 rows
    char* ws = (char*)d_ws;
    size_t off = 0;
    auto alloc = [&](size_t bytes) -> char* {
        char* p = ws + off;
        off += (bytes + 255) & ~(size_t)255;
        return p;
    };
    _Float16* xh   = (_Float16*)alloc(MS * EMB * 2);                 // 16.8 MB
    _Float16* BqT  = (_Float16*)alloc((size_t)EMB * EMB * 2);        //  8.4 MB
    _Float16* BkT  = (_Float16*)alloc((size_t)EMB * NKV * HD * 2);   //  2.1 MB
    _Float16* BvT  = (_Float16*)alloc((size_t)EMB * NKV * HD * 2);   //  2.1 MB
    _Float16* BoT  = (_Float16*)alloc((size_t)EMB * EMB * 2);        //  8.4 MB
    float*    qf32 = (float*)alloc(MS * EMB * 4);                    // 33.5 MB
    float*    kf32 = (float*)alloc(MS * NKV * HD * 4);               //  8.4 MB
    float*    vf32 = (float*)alloc(MS * NKV * HD * 4);               //  8.4 MB
    _Float16* qh   = (_Float16*)alloc(MS * EMB * 2);                 // 16.8 MB
    _Float16* kTh  = (_Float16*)alloc(MS * NKV * HD * 2);            //  4.2 MB
    _Float16* vh   = (_Float16*)alloc(MS * NKV * HD * 2);            //  4.2 MB
    _Float16* atth = (_Float16*)qf32;  // alias: qf32 dead after rope_q

    // 1) stage inputs to f16
    f32_to_f16_kernel<<<(MS * EMB) / 256, 256, 0, stream>>>(x, xh, (int)(MS * EMB));
    transpose_f16_kernel<<<(EMB * EMB) / 256, 256, 0, stream>>>(Wq, BqT, EMB, EMB);
    transpose_f16_kernel<<<(EMB * NKV * HD) / 256, 256, 0, stream>>>(Wk, BkT, NKV * HD, EMB);
    transpose_f16_kernel<<<(EMB * NKV * HD) / 256, 256, 0, stream>>>(Wv, BvT, NKV * HD, EMB);
    transpose_f16_kernel<<<(EMB * EMB) / 256, 256, 0, stream>>>(Wo, BoT, EMB, EMB);

    // 2) projections (WMMA)
    gemm_wmma_kernel<<<dim3(EMB / 64, MS / 256), 256, 0, stream>>>(xh, BqT, qf32, (int)MS, EMB, EMB);
    gemm_wmma_kernel<<<dim3((NKV * HD) / 64, MS / 256), 256, 0, stream>>>(xh, BkT, kf32, (int)MS, NKV * HD, EMB);
    gemm_wmma_kernel<<<dim3((NKV * HD) / 64, MS / 256), 256, 0, stream>>>(xh, BvT, vf32, (int)MS, NKV * HD, EMB);

    // 3) RoPE + attention-friendly f16 layouts
    rope_q_kernel<<<(BATCH * SLEN * NH * 32) / 256, 256, 0, stream>>>(qf32, cosT, sinT, qh);
    rope_k_kernel<<<(BATCH * SLEN * NKV * 32) / 256, 256, 0, stream>>>(kf32, cosT, sinT, kTh);
    conv_v_kernel<<<(BATCH * SLEN * NKV * HD) / 256, 256, 0, stream>>>(vf32, vh);

    // 4) flash attention (WMMA)
    attn_wmma_kernel<<<(BATCH * NKV * GRP * (SLEN / 16)) / 8, 256, 0, stream>>>(qh, kTh, vh, atth);

    // 5) output projection (WMMA) -> f32 d_out
    gemm_wmma_kernel<<<dim3(EMB / 64, MS / 256), 256, 0, stream>>>(atth, BoT, out, (int)MS, EMB, EMB);
}
